// TransConvLayer_60610578481395
// MI455X (gfx1250) — matrix operand order repaired and checked by reading, not compiled, not run on hardware
//
#include <hip/hip_runtime.h>
#include <hip/hip_bf16.h>

typedef __bf16 bf16;
typedef __attribute__((ext_vector_type(16))) bf16 bf16x16;
typedef __attribute__((ext_vector_type(8)))  bf16 bf16x8;
typedef __attribute__((ext_vector_type(8)))  float f32x8;
typedef __attribute__((ext_vector_type(2)))  float f32x2;

#define B_  8
#define N_  16384
#define C_  256
#define HD_ 512
#define H_  8
#define D_  64
#define TL_ 128

// ---------- helpers ----------

__device__ __forceinline__ bf16 f2bf(float f) {
  union { float f; unsigned u; } v; v.f = f;
  unsigned r = v.u + 0x7FFFu + ((v.u >> 16) & 1u);   // round-nearest-even
  unsigned short h = (unsigned short)(r >> 16);
  bf16 b; __builtin_memcpy(&b, &h, sizeof(b));
  return b;
}

__device__ __forceinline__ f32x8 zero8() {
  return (f32x8){0.f,0.f,0.f,0.f,0.f,0.f,0.f,0.f};
}

__device__ __forceinline__ f32x8 wmma_bf16(bf16x16 a, bf16x16 b, f32x8 c) {
  // D = A(16x32 bf16) * B(32x16 bf16) + C(16x16 f32)
  return __builtin_amdgcn_wmma_f32_16x16x32_bf16(false, a, false, b, (short)0, c, false, false);
}

// A tile (16x32) from bf16 row-major (LDS), leading dim ld.
// ISA layout: lane m (0..15): K = {0..7, 16..23}; lane m+16: K = {8..15, 24..31}
__device__ __forceinline__ bf16x16 ldA_bf16(const bf16* T, int ld, int row0, int k0,
                                            int m15, int hi) {
  const bf16* p = T + (size_t)(row0 + m15) * ld + k0 + hi * 8;
  bf16x8 c0 = *(const bf16x8*)p;
  bf16x8 c1 = *(const bf16x8*)(p + 16);
  bf16x16 a;
#pragma unroll
  for (int i = 0; i < 8; ++i) { a[i] = c0[i]; a[i + 8] = c1[i]; }
  return a;
}

// B tile (32x16) from bf16 where storage is Bt[col][k] row-major with leading dim ld.
// ISA layout: lane n (0..15): col n, K = {0..7, 16..23}; lane n+16: col n, K = {8..15, 24..31}
__device__ __forceinline__ bf16x16 ldB_bf16(const bf16* Bt, int ld, int col0, int k0,
                                            int m15, int hi) {
  const bf16* p = Bt + (size_t)(col0 + m15) * ld + k0 + hi * 8;
  bf16x8 c0 = *(const bf16x8*)p;
  bf16x8 c1 = *(const bf16x8*)(p + 16);
  bf16x16 b;
#pragma unroll
  for (int i = 0; i < 8; ++i) { b[i] = c0[i]; b[i + 8] = c1[i]; }
  return b;
}

// Cooperative stage of a fp32 [TL x C] activation tile into bf16 LDS (row-major).
__device__ __forceinline__ void stage_X(const float* __restrict__ X, bf16* dst, int tid) {
  for (int i = tid; i < TL_ * C_ / 8; i += 256) {
    f32x8 v = *(const f32x8*)(X + (size_t)i * 8);
    bf16x8 o;
#pragma unroll
    for (int j = 0; j < 8; ++j) o[j] = f2bf(v[j]);
    *(bf16x8*)(dst + (size_t)i * 8) = o;
  }
}

// ---------- kernel 0: transpose-convert weights fp32[C][HD] -> bf16[HD][C] ----------

__global__ void wt_prep_kernel(const float* __restrict__ Wq, const float* __restrict__ Wk,
                               const float* __restrict__ Wv,
                               bf16* __restrict__ Wqt, bf16* __restrict__ Wkt,
                               bf16* __restrict__ Wvt) {
  const int j   = blockIdx.x;   // output column 0..511
  const int mat = blockIdx.y;   // 0..2
  const float* W = (mat == 0) ? Wq : (mat == 1) ? Wk : Wv;
  bf16* Wt       = (mat == 0) ? Wqt : (mat == 1) ? Wkt : Wvt;
  const int i = threadIdx.x;    // 0..255 (== C)
  Wt[(size_t)j * C_ + i] = f2bf(W[(size_t)i * HD_ + j]);
}

// ---------- kernel 1: K/V projection + normalize + kv / ksum / vsum accumulation ----
// Wave w owns rows [w*16, w*16+16) and sweeps all heads (columns). The only
// loop-invariant operand is the wave's own 16x256 A slice, held in 64 VGPRs.

__global__ void kv_kernel(const float* __restrict__ Xs,
                          const bf16* __restrict__ Wkt, const bf16* __restrict__ Wvt,
                          const float* __restrict__ bk, const float* __restrict__ bv,
                          float* __restrict__ kvs, float* __restrict__ ksum,
                          float* __restrict__ vsum) {
  extern __shared__ char smem[];
  bf16* Xl   = (bf16*)smem;            // [TL][C]  bf16, 64 KB (shared A operand)
  bf16* ldsK = Xl + (size_t)TL_ * C_;  // [H][D][TL] : K^T (normalized), bf16, 128 KB
  bf16* ldsV = ldsK + H_ * D_ * TL_;   // [H][D][TL] : V^T, bf16, 128 KB

  const int lane = threadIdx.x & 31;
  const int w    = threadIdx.x >> 5;   // wave == row-chunk
  const int m15  = lane & 15;
  const int hi   = lane >> 4;
  const int b    = blockIdx.y;
  const int l0   = blockIdx.x * TL_;
  const float* X = Xs + ((size_t)b * N_ + l0) * C_;

  stage_X(X, Xl, threadIdx.x);
  __syncthreads();

  // Hold this wave's A rows (16 x 256 bf16 = 64 VGPRs) in registers.
  bf16x16 Ak[8];
#pragma unroll
  for (int k = 0; k < 8; ++k) Ak[k] = ldA_bf16(Xl, C_, w * 16, k * 32, m15, hi);

#pragma unroll 1
  for (int h = 0; h < H_; ++h) {
    const int col0 = h * D_;
    // ======== K ========
    {
      f32x8 acc[4];
#pragma unroll
      for (int t = 0; t < 4; ++t) acc[t] = zero8();
#pragma unroll
      for (int k = 0; k < 8; ++k)
#pragma unroll
        for (int t = 0; t < 4; ++t)
          acc[t] = wmma_bf16(Ak[k], ldB_bf16(Wkt, C_, col0 + t * 16, k * 32, m15, hi), acc[t]);

#pragma unroll
      for (int t = 0; t < 4; ++t) {
        const float bb = bk[col0 + t * 16 + m15];
#pragma unroll
        for (int g = 0; g < 8; ++g) acc[t][g] += bb;
      }
      // L2-normalize rows. C/D layout: VGPR g, half hi -> row (g + 8*hi), col = t*16 + m15
#pragma unroll
      for (int g = 0; g < 8; ++g) {
        float ss = 0.f;
#pragma unroll
        for (int t = 0; t < 4; ++t) ss += acc[t][g] * acc[t][g];
        ss += __shfl_xor(ss, 1, 16);
        ss += __shfl_xor(ss, 2, 16);
        ss += __shfl_xor(ss, 4, 16);
        ss += __shfl_xor(ss, 8, 16);
        float rn = rsqrtf(ss + 1e-20f);
#pragma unroll
        for (int t = 0; t < 4; ++t) acc[t][g] *= rn;
      }
      bf16* Kh = ldsK + (size_t)h * D_ * TL_;
#pragma unroll
      for (int t = 0; t < 4; ++t) {
        float sk = 0.f;
        const int d = t * 16 + m15;
#pragma unroll
        for (int g = 0; g < 8; ++g) {
          sk += acc[t][g];
          Kh[(size_t)d * TL_ + w * 16 + g + hi * 8] = f2bf(acc[t][g]);
        }
        sk += __shfl_xor(sk, 16, 32);
        if (hi == 0)
          atomicAdd(&ksum[((size_t)b * H_ + h) * D_ + d], sk);
      }
    }
    // ======== V ========
    {
      f32x8 acc[4];
#pragma unroll
      for (int t = 0; t < 4; ++t) acc[t] = zero8();
#pragma unroll
      for (int k = 0; k < 8; ++k)
#pragma unroll
        for (int t = 0; t < 4; ++t)
          acc[t] = wmma_bf16(Ak[k], ldB_bf16(Wvt, C_, col0 + t * 16, k * 32, m15, hi), acc[t]);

      bf16* Vh = ldsV + (size_t)h * D_ * TL_;
#pragma unroll
      for (int t = 0; t < 4; ++t) {
        const float bb = bv[col0 + t * 16 + m15];
        float sv = 0.f;
        const int d = t * 16 + m15;
#pragma unroll
        for (int g = 0; g < 8; ++g) {
          float v = acc[t][g] + bb;
          sv += v;
          Vh[(size_t)d * TL_ + w * 16 + g + hi * 8] = f2bf(v);
        }
        sv += __shfl_xor(sv, 16, 32);
        if (hi == 0)
          atomicAdd(&vsum[((size_t)b * H_ + h) * D_ + d], sv);
      }
    }
  }

  __syncthreads();

  // ---- kv = K^T (64 x 128) @ V (128 x 64); wave == head here ----
  float* kvdst = kvs + ((size_t)b * H_ + w) * D_ * D_;
  const bf16* Kh = ldsK + (size_t)w * D_ * TL_;
  const bf16* Vh = ldsV + (size_t)w * D_ * TL_;
#pragma unroll 1
  for (int mt = 0; mt < 4; ++mt) {
#pragma unroll
    for (int nt = 0; nt < 4; ++nt) {
      f32x8 acc = zero8();
#pragma unroll
      for (int k = 0; k < 4; ++k) {
        bf16x16 A  = ldA_bf16(Kh, TL_, mt * 16, k * 32, m15, hi);
        bf16x16 Bv = ldB_bf16(Vh, TL_, nt * 16, k * 32, m15, hi);
        acc = wmma_bf16(A, Bv, acc);
      }
#pragma unroll
      for (int g = 0; g < 8; ++g)
        atomicAdd(&kvdst[(size_t)(mt * 16 + g + hi * 8) * D_ + nt * 16 + m15], acc[g]);
    }
  }
}

// ---------- kernel 2: Q projection + normalize + q@kv (fp32 WMMA) + output ----------
// Wave w owns rows [w*16, w*16+16) and loops over heads.

__global__ void q_out_kernel(const float* __restrict__ Xq,
                             const bf16* __restrict__ Wqt, const float* __restrict__ bq,
                             const float* __restrict__ kvs, const float* __restrict__ ksum,
                             const float* __restrict__ vsum, float* __restrict__ out) {
  extern __shared__ char smem[];
  bf16*  Xl   = (bf16*)smem;                      // [TL][C] bf16, 64 KB
  float* kvh  = (float*)(Xl + (size_t)TL_ * C_);  // [H][D][D] f32, 128 KB
  float* oacc = kvh + H_ * D_ * D_;               // [TL][D]  f32, 32 KB
  float* qst  = oacc + TL_ * D_;                  // [8 waves][16][D] f32, 32 KB

  const int lane = threadIdx.x & 31;
  const int w    = threadIdx.x >> 5;
  const int m15  = lane & 15;
  const int hi   = lane >> 4;
  const int b    = blockIdx.y;
  const int l0   = blockIdx.x * TL_;
  const float* X = Xq + ((size_t)b * N_ + l0) * C_;

  stage_X(X, Xl, threadIdx.x);

  // stage all heads' kv blocks into LDS (fp32), cooperatively
  {
    const float4* src = (const float4*)(kvs + (size_t)b * H_ * D_ * D_);
    float4* dst = (float4*)kvh;
    for (int i = threadIdx.x; i < H_ * D_ * D_ / 4; i += 256) dst[i] = src[i];
  }
  // zero the output accumulator
  for (int i = threadIdx.x; i < TL_ * D_; i += 256) oacc[i] = 0.f;
  __syncthreads();

  // Hold this wave's A rows in registers (64 VGPRs).
  bf16x16 Ak[8];
#pragma unroll
  for (int k = 0; k < 8; ++k) Ak[k] = ldA_bf16(Xl, C_, w * 16, k * 32, m15, hi);

  float* qs = qst + (size_t)w * 16 * D_;          // per-wave q staging (16 x 64)

#pragma unroll 1
  for (int h = 0; h < H_; ++h) {
    const int col0 = h * D_;
    // ---- Q projection for this head's 64 columns ----
    f32x8 aQ[4];
#pragma unroll
    for (int t = 0; t < 4; ++t) aQ[t] = zero8();
#pragma unroll
    for (int k = 0; k < 8; ++k)
#pragma unroll
      for (int t = 0; t < 4; ++t)
        aQ[t] = wmma_bf16(Ak[k], ldB_bf16(Wqt, C_, col0 + t * 16, k * 32, m15, hi), aQ[t]);

    float ksr[4];
#pragma unroll
    for (int t = 0; t < 4; ++t) {
      const float bb = bq[col0 + t * 16 + m15];
      ksr[t] = ksum[((size_t)b * H_ + h) * D_ + t * 16 + m15];
#pragma unroll
      for (int g = 0; g < 8; ++g) aQ[t][g] += bb;
    }

    // ---- normalize + denominator (q . ksum) + stage q rows to LDS ----
    float denomv[8];
#pragma unroll
    for (int g = 0; g < 8; ++g) {
      float ss = 0.f;
#pragma unroll
      for (int t = 0; t < 4; ++t) ss += aQ[t][g] * aQ[t][g];
      ss += __shfl_xor(ss, 1, 16);
      ss += __shfl_xor(ss, 2, 16);
      ss += __shfl_xor(ss, 4, 16);
      ss += __shfl_xor(ss, 8, 16);
      float rn = rsqrtf(ss + 1e-20f);
      float dd = 0.f;
#pragma unroll
      for (int t = 0; t < 4; ++t) {
        aQ[t][g] *= rn;
        dd += aQ[t][g] * ksr[t];
        qs[(size_t)(g + hi * 8) * D_ + t * 16 + m15] = aQ[t][g];
      }
      dd += __shfl_xor(dd, 1, 16);
      dd += __shfl_xor(dd, 2, 16);
      dd += __shfl_xor(dd, 4, 16);
      dd += __shfl_xor(dd, 8, 16);
      denomv[g] = dd + 32768.0f;                   // + N + N
    }

    // ---- num = q (16x64) @ kv (64x64) via V_WMMA_F32_16X16X4_F32 (full fp32) ----
    const float* kp = kvh + (size_t)h * D_ * D_;
#pragma unroll
    for (int nt = 0; nt < 4; ++nt) {
      const float vsr = vsum[((size_t)b * H_ + h) * D_ + nt * 16 + m15];
      f32x8 acc = zero8();
#pragma unroll
      for (int k = 0; k < 16; ++k) {
        // A (16x4 f32): lane m: K = 2*hi, 2*hi+1
        f32x2 a;
        a.x = qs[(size_t)m15 * D_ + k * 4 + 2 * hi];
        a.y = qs[(size_t)m15 * D_ + k * 4 + 2 * hi + 1];
        // B (4x16 f32): lane n: rows K = 2*hi, 2*hi+1, column n
        f32x2 bb;
        bb.x = kp[(size_t)(k * 4 + 2 * hi) * D_ + nt * 16 + m15];
        bb.y = kp[(size_t)(k * 4 + 2 * hi + 1) * D_ + nt * 16 + m15];
        acc = __builtin_amdgcn_wmma_f32_16x16x4_f32(false, a, false, bb, (short)0, acc,
                                                    false, false);
      }
      // epilogue: (num + N*vsum) / denom, mean over heads via ds_add_f32
#pragma unroll
      for (int g = 0; g < 8; ++g) {
        float val = (acc[g] + 16384.0f * vsr) / denomv[g] * 0.125f;
        atomicAdd(&oacc[(size_t)(w * 16 + g + hi * 8) * D_ + nt * 16 + m15], val);
      }
    }
  }

  __syncthreads();

  // coalesced write-out: [TL][D] f32 -> out[b][l0 : l0+TL][0:64]
  const float4* oa4 = (const float4*)oacc;
  float4* o4 = (float4*)(out + ((size_t)b * N_ + l0) * D_);
  for (int i = threadIdx.x; i < TL_ * D_ / 4; i += 256) o4[i] = oa4[i];
}

// ---------- host launcher ----------

extern "C" void kernel_launch(void* const* d_in, const int* in_sizes, int n_in,
                              void* d_out, int out_size, void* d_ws, size_t ws_size,
                              hipStream_t stream) {
  (void)in_sizes; (void)n_in; (void)out_size; (void)ws_size;
  const float* qin = (const float*)d_in[0];
  const float* sin = (const float*)d_in[1];
  const float* Wq  = (const float*)d_in[2];
  const float* bq  = (const float*)d_in[3];
  const float* Wk  = (const float*)d_in[4];
  const float* bk  = (const float*)d_in[5];
  const float* Wv  = (const float*)d_in[6];
  const float* bv  = (const float*)d_in[7];
  float* out = (float*)d_out;

  // workspace carve-out (~1.8 MB)
  char* ws = (char*)d_ws;
  bf16* Wqt = (bf16*)ws;
  bf16* Wkt = Wqt + (size_t)HD_ * C_;
  bf16* Wvt = Wkt + (size_t)HD_ * C_;
  float* kvs  = (float*)(Wvt + (size_t)HD_ * C_);
  float* ksum = kvs + (size_t)B_ * H_ * D_ * D_;
  float* vsum = ksum + (size_t)B_ * H_ * D_;

  const size_t accBytes = ((size_t)B_ * H_ * D_ * D_ + 2u * B_ * H_ * D_) * sizeof(float);
  hipMemsetAsync(kvs, 0, accBytes, stream);

  wt_prep_kernel<<<dim3(HD_, 3), C_, 0, stream>>>(Wq, Wk, Wv, Wqt, Wkt, Wvt);

  // 64 KB X-stage + 2 * 128 KB K^T/V^T = 320 KB (CDNA5 per-WG max)
  const size_t smem1 = ((size_t)TL_ * C_ + 2u * H_ * D_ * TL_) * sizeof(bf16);
  kv_kernel<<<dim3(N_ / TL_, B_), 256, smem1, stream>>>(sin, Wkt, Wvt, bk, bv,
                                                        kvs, ksum, vsum);

  // 64 KB X-stage + 128 KB kv + 32 KB out-acc + 32 KB q-stage = 256 KB
  const size_t smem2 = (size_t)TL_ * C_ * sizeof(bf16) +
                       (size_t)(H_ * D_ * D_ + TL_ * D_ + H_ * 16 * D_) * sizeof(float);
  q_out_kernel<<<dim3(N_ / TL_, B_), 256, smem2, stream>>>(qin, Wqt, bq,
                                                           kvs, ksum, vsum, out);
}
